// CharAttention_89180700934421
// MI455X (gfx1250) — matrix operand order repaired and checked by reading, not compiled, not run on hardware
//
#include <hip/hip_runtime.h>
#include <cstdint>

// ---------------------------------------------------------------------------
// CharAttention on MI455X (gfx1250, wave32, WMMA).
//   words = x[mask]; q = words @ Wq^T; kv = x @ Wkv^T; k,v = split(kv)
//   S = q @ k^T (causal-masked softmax, f32) ; O = att @ v
//   out = O @ Wproj^T + words ; x_new = x (+)scatter-add O at mask rows
// GEMMs: bf16 operands, f32 WMMA accumulation (v_wmma_f32_16x16x32_bf16),
// 64x64 wave tiles, double-buffered LDS filled with CDNA5 async LDS-direct
// loads (global_load_async_to_lds_b128 + s_wait_asynccnt): no VGPR staging,
// no spills. Intermediates (~120MB) live in the 192MB L2 -> staged pipeline
// instead of flash fusion; the chain is compute-bound on the matrix units.
// Workspace: ~169 MB.
// ---------------------------------------------------------------------------

#define SEQ 4096
#define NE  1024

typedef __bf16 bf16_t;
typedef __attribute__((ext_vector_type(16))) __bf16 v16bf;
typedef __attribute__((ext_vector_type(8)))  float  v8f;

// CDNA5 async global->LDS DMA: each lane moves 16B, tracked by ASYNCcnt.
// ldsAddr = wave-relative LDS byte offset (= low 32 bits of flat shared ptr).
__device__ __forceinline__ void async_ld_b128(uint32_t ldsAddr, const void* gaddr) {
  asm volatile("global_load_async_to_lds_b128 %0, %1, off"
               :: "v"(ldsAddr), "v"(gaddr) : "memory");
}
__device__ __forceinline__ void wait_asynccnt0() {
  asm volatile("s_wait_asynccnt 0x0" ::: "memory");
}

// ---------------- elementwise helpers ----------------

__global__ void k_f32_to_bf16(const float* __restrict__ in, bf16_t* __restrict__ out, long n) {
  long i = (long)blockIdx.x * blockDim.x + threadIdx.x;
  long stride = (long)gridDim.x * blockDim.x;
  for (; i < n; i += stride) out[i] = (bf16_t)in[i];
}

__global__ void k_copy_f32(const float* __restrict__ in, float* __restrict__ out, long n) {
  long i = (long)blockIdx.x * blockDim.x + threadIdx.x;
  long stride = (long)gridDim.x * blockDim.x;
  for (; i < n; i += stride) out[i] = in[i];
}

// x_new[mask[r], :] += attn_out[r, :]  (atomic to honor duplicate indices)
__global__ void k_scatter_add(const float* __restrict__ src, const int* __restrict__ idx,
                              float* __restrict__ dst, int rows, int cols) {
  long i = (long)blockIdx.x * blockDim.x + threadIdx.x;
  if (i >= (long)rows * cols) return;
  int r = (int)(i / cols);
  int c = (int)(i - (long)r * cols);
  atomicAdd(&dst[(long)idx[r] * cols + c], src[i]);
}

// 32x32 LDS-tiled bf16 transpose (builds v^T so att@v becomes an NT GEMM)
__global__ __launch_bounds__(256)
void k_transpose_bf16(const bf16_t* __restrict__ src, long lds,
                      bf16_t* __restrict__ dst, long ldd) {
  __shared__ bf16_t tile[32][33];
  int c0 = blockIdx.x * 32;
  int r0 = blockIdx.y * 32;
  int tx = threadIdx.x & 31;
  int ty = threadIdx.x >> 5;  // 0..7
#pragma unroll
  for (int i = 0; i < 32; i += 8)
    tile[ty + i][tx] = src[(long)(r0 + ty + i) * lds + c0 + tx];
  __syncthreads();
#pragma unroll
  for (int i = 0; i < 32; i += 8)
    dst[(long)(c0 + ty + i) * ldd + r0 + tx] = tile[tx][ty + i];
}

// ---------------- generic bf16 NT GEMM: C[M,N] = A[MxK] * B[NxK]^T ----------------
// 128 threads = 4 wave32s (2x2); block tile 128x128, BK=32, double-buffered LDS
// filled by async LDS-direct loads. Wave tile 64x64 = 4x4 WMMA accumulators.
// Optional: row-gather on A (words = x[mask]), f32 epilogue add with gather
// (out += words), bf16 and/or f32 outputs, causal block skip via mask max.

__global__ __launch_bounds__(128, 1)
void k_gemm_nt(const bf16_t* __restrict__ A, long lda, const int* __restrict__ gatherA,
               const bf16_t* __restrict__ B, long ldb,
               float* __restrict__ Cf, bf16_t* __restrict__ Cb, long ldc,
               int M, int N, int K,
               const float* __restrict__ addRows, const int* __restrict__ addGather, long addLd,
               const int* __restrict__ causalMask) {
  constexpr int BM = 128, BN = 128, BK = 32, LDSW = BK + 8;  // pad: conflict-free b128 reads
  __shared__ bf16_t As[2][BM * LDSW];
  __shared__ bf16_t Bs[2][BN * LDSW];
  __shared__ int sMax;

  const int tid = threadIdx.x;
  const int m0 = blockIdx.y * BM;
  const int n0 = blockIdx.x * BN;

  // Skip score blocks that are entirely in the masked (future) region.
  if (causalMask) {
    if (tid == 0) sMax = -1;
    __syncthreads();
    atomicMax(&sMax, causalMask[m0 + tid]);   // 128 threads cover the 128 rows
    __syncthreads();
    if (n0 > sMax) return;  // uniform exit: EXEC stays full for WMMA
  }

  const int lane = tid & 31;
  const int wave = tid >> 5;      // 0..3
  const int waveM = wave >> 1;    // 0..1 : 64 rows
  const int waveN = wave & 1;     // 0..1 : 64 cols

  // global -> LDS staging: one thread per tile row, 64B (32 bf16) per row,
  // moved with 4 async b128 DMAs (no destination VGPRs).
  const long aRow = gatherA ? (long)gatherA[m0 + tid] : (long)(m0 + tid);
  const bf16_t* aPtr = A + aRow * lda;
  const bf16_t* bPtr = B + (long)(n0 + tid) * ldb;
  // wave-relative LDS byte offsets (flat shared ptr low 32 bits, ISA 10.2)
  uint32_t aLds[2], bLds[2];
  aLds[0] = (uint32_t)(uintptr_t)&As[0][tid * LDSW];
  aLds[1] = (uint32_t)(uintptr_t)&As[1][tid * LDSW];
  bLds[0] = (uint32_t)(uintptr_t)&Bs[0][tid * LDSW];
  bLds[1] = (uint32_t)(uintptr_t)&Bs[1][tid * LDSW];

  // WMMA fragment addressing (ISA 7.12.2 VGPR layouts, wave32):
  //  A 16x32: lane<16 holds row(lane) K={kc..kc+7, kc+16..kc+23}, kc=0; lane>=16: kc=8
  //  B 32x16: lane<16 holds col(lane) K={kc..kc+7, kc+16..kc+23}, kc=0; lane>=16: kc=8
  const int fr  = lane & 15;
  const int aKc = (lane >> 4) * 8;
  const int bKc = (lane >> 4) * 8;

  v8f acc[4][4];
#pragma unroll
  for (int i = 0; i < 4; i++)
#pragma unroll
    for (int j = 0; j < 4; j++)
      acc[i][j] = (v8f){0.f, 0.f, 0.f, 0.f, 0.f, 0.f, 0.f, 0.f};

  // prologue: async-stage k-tile 0 into buffer 0
#pragma unroll
  for (int i = 0; i < 4; i++) {
    async_ld_b128(aLds[0] + i * 16, aPtr + i * 8);
    async_ld_b128(bLds[0] + i * 16, bPtr + i * 8);
  }
  wait_asynccnt0();
  __syncthreads();

  int buf = 0;
  for (int k0 = 0; k0 < K; k0 += BK) {
    const bool hasNext = (k0 + BK) < K;
    if (hasNext) {  // DMA next tile into the other buffer under the WMMA burst
#pragma unroll
      for (int i = 0; i < 4; i++) {
        async_ld_b128(aLds[buf ^ 1] + i * 16, aPtr + k0 + BK + i * 8);
        async_ld_b128(bLds[buf ^ 1] + i * 16, bPtr + k0 + BK + i * 8);
      }
    }

    const bf16_t* aBuf = As[buf];
    const bf16_t* bBuf = Bs[buf];
    v16bf aF[4], bF[4];
#pragma unroll
    for (int mt = 0; mt < 4; mt++) {
      const bf16_t* p = aBuf + (waveM * 64 + mt * 16 + fr) * LDSW + aKc;
      ((uint4*)&aF[mt])[0] = *(const uint4*)p;
      ((uint4*)&aF[mt])[1] = *(const uint4*)(p + 16);
    }
#pragma unroll
    for (int nt = 0; nt < 4; nt++) {
      const bf16_t* p = bBuf + (waveN * 64 + nt * 16 + fr) * LDSW + bKc;
      ((uint4*)&bF[nt])[0] = *(const uint4*)p;
      ((uint4*)&bF[nt])[1] = *(const uint4*)(p + 16);
    }
#pragma unroll
    for (int mt = 0; mt < 4; mt++)
#pragma unroll
      for (int nt = 0; nt < 4; nt++)
        acc[mt][nt] = __builtin_amdgcn_wmma_f32_16x16x32_bf16(
            false, aF[mt], false, bF[nt], (short)0, acc[mt][nt], false, false);

    if (hasNext) {
      wait_asynccnt0();   // own DMAs into buf^1 landed
      __syncthreads();    // everyone's DMAs visible; readers of buf done
      buf ^= 1;
    }
  }

  // C/D layout: vgpr r -> row (r + 8*(lane>=16)), col = lane&15
  const int colBase = n0 + waveN * 64 + fr;
  const int rowSub  = (lane >> 4) * 8;
#pragma unroll
  for (int mt = 0; mt < 4; mt++) {
    int rowBase = m0 + waveM * 64 + mt * 16 + rowSub;
#pragma unroll
    for (int nt = 0; nt < 4; nt++) {
      int col = colBase + nt * 16;
#pragma unroll
      for (int r = 0; r < 8; r++) {
        int row = rowBase + r;
        float v = acc[mt][nt][r];
        if (addRows) {
          long ar = addGather ? (long)addGather[row] : (long)row;
          v += addRows[ar * addLd + col];
        }
        long off = (long)row * ldc + col;
        if (Cf) Cf[off] = v;
        if (Cb) Cb[off] = (bf16_t)v;
      }
    }
  }
}

// ---------------- causal softmax (f32 scores -> bf16 att), one block per row ----------------

__global__ __launch_bounds__(256)
void k_softmax_causal(const float* __restrict__ scores, const int* __restrict__ mask,
                      bf16_t* __restrict__ att, int T) {
  __shared__ float sred[8];
  const int r = blockIdx.x, tid = threadIdx.x;
  const int limit = mask[r];
  const float* row = scores + (long)r * T;

  float vals[16];
  float mx = -INFINITY;
#pragma unroll
  for (int i = 0; i < 16; i++) {
    int j = tid + i * 256;
    float s = (j <= limit) ? row[j] : -INFINITY;
    vals[i] = s;
    mx = fmaxf(mx, s);
  }
  for (int o = 16; o > 0; o >>= 1) mx = fmaxf(mx, __shfl_xor(mx, o, 32));
  if ((tid & 31) == 0) sred[tid >> 5] = mx;
  __syncthreads();
  mx = sred[0];
#pragma unroll
  for (int i = 1; i < 8; i++) mx = fmaxf(mx, sred[i]);
  __syncthreads();

  float sum = 0.f;
#pragma unroll
  for (int i = 0; i < 16; i++) {
    float e = (vals[i] > -INFINITY) ? __expf(vals[i] - mx) : 0.f;
    vals[i] = e;
    sum += e;
  }
  for (int o = 16; o > 0; o >>= 1) sum += __shfl_xor(sum, o, 32);
  if ((tid & 31) == 0) sred[tid >> 5] = sum;
  __syncthreads();
  sum = 0.f;
#pragma unroll
  for (int i = 0; i < 8; i++) sum += sred[i];

  float inv = (sum > 0.f) ? 1.f / sum : 0.f;
  bf16_t* arow = att + (long)r * T;
#pragma unroll
  for (int i = 0; i < 16; i++) arow[tid + i * 256] = (bf16_t)(vals[i] * inv);
}

// ---------------- driver ----------------

extern "C" void kernel_launch(void* const* d_in, const int* in_sizes, int n_in,
                              void* d_out, int out_size, void* d_ws, size_t ws_size,
                              hipStream_t stream) {
  const float* x     = (const float*)d_in[0];
  const int*   mask  = (const int*)d_in[1];
  const float* Wq    = (const float*)d_in[2];
  const float* Wkv   = (const float*)d_in[3];
  const float* Wproj = (const float*)d_in[4];
  float* out  = (float*)d_out;
  float* xnew = out + (size_t)SEQ * NE;

  size_t off = 0;
  auto alloc = [&](size_t bytes) -> void* {
    void* p = (char*)d_ws + off;
    off += (bytes + 255) & ~(size_t)255;
    return p;
  };
  bf16_t* xb    = (bf16_t*)alloc((size_t)SEQ * NE * 2);        //  8 MB
  bf16_t* Wqb   = (bf16_t*)alloc((size_t)NE * NE * 2);         //  2 MB
  bf16_t* Wkvb  = (bf16_t*)alloc((size_t)2 * NE * NE * 2);     //  4 MB
  bf16_t* Wpb   = (bf16_t*)alloc((size_t)NE * NE * 2);         //  2 MB
  bf16_t* qb    = (bf16_t*)alloc((size_t)SEQ * NE * 2);        //  8 MB
  bf16_t* kvb   = (bf16_t*)alloc((size_t)SEQ * 2 * NE * 2);    // 16 MB
  bf16_t* vT    = (bf16_t*)alloc((size_t)NE * SEQ * 2);        //  8 MB
  bf16_t* attnb = (bf16_t*)alloc((size_t)SEQ * NE * 2);        //  8 MB
  float*  attnf = (float*)alloc((size_t)SEQ * NE * 4);         // 16 MB
  float*  scor  = (float*)alloc((size_t)SEQ * SEQ * 4);        // 64 MB
  bf16_t* attb  = (bf16_t*)alloc((size_t)SEQ * SEQ * 2);       // 32 MB

  // downconvert inputs to bf16 (WMMA operand precision)
  k_f32_to_bf16<<<2048, 256, 0, stream>>>(x, xb, (long)SEQ * NE);
  k_f32_to_bf16<<<512, 256, 0, stream>>>(Wq, Wqb, (long)NE * NE);
  k_f32_to_bf16<<<1024, 256, 0, stream>>>(Wkv, Wkvb, (long)2 * NE * NE);
  k_f32_to_bf16<<<512, 256, 0, stream>>>(Wproj, Wpb, (long)NE * NE);

  // q = x[mask] @ Wq^T   (row gather folded into A addressing)
  k_gemm_nt<<<dim3(NE / 128, SEQ / 128), 128, 0, stream>>>(
      xb, NE, mask, Wqb, NE, nullptr, qb, NE, SEQ, NE, NE, nullptr, nullptr, 0, nullptr);

  // kv = x @ Wkv^T  (k = cols [0,NE), v = cols [NE,2NE))
  k_gemm_nt<<<dim3(2 * NE / 128, SEQ / 128), 128, 0, stream>>>(
      xb, NE, nullptr, Wkvb, NE, nullptr, kvb, 2 * NE, SEQ, 2 * NE, NE, nullptr, nullptr, 0, nullptr);

  // v^T so att@v is an NT GEMM
  k_transpose_bf16<<<dim3(NE / 32, SEQ / 32), 256, 0, stream>>>(kvb + NE, 2 * NE, vT, SEQ);

  // scores = q @ k^T (f32 out, lives in L2); fully-masked blocks skipped
  k_gemm_nt<<<dim3(SEQ / 128, SEQ / 128), 128, 0, stream>>>(
      qb, NE, nullptr, kvb, 2 * NE, scor, nullptr, SEQ, SEQ, SEQ, NE, nullptr, nullptr, 0, mask);

  // att = softmax(mask(scores)) -> bf16
  k_softmax_causal<<<SEQ, 256, 0, stream>>>(scor, mask, attb, SEQ);

  // attn_out = att @ v  (f32 for x_new scatter, bf16 for proj GEMM)
  k_gemm_nt<<<dim3(NE / 128, SEQ / 128), 128, 0, stream>>>(
      attb, SEQ, nullptr, vT, SEQ, attnf, attnb, NE, SEQ, NE, SEQ, nullptr, nullptr, 0, nullptr);

  // out = attn_out @ Wproj^T + words   (words add via gathered epilogue)
  k_gemm_nt<<<dim3(NE / 128, SEQ / 128), 128, 0, stream>>>(
      attnb, NE, nullptr, Wpb, NE, out, nullptr, NE, SEQ, NE, NE, x, mask, NE, nullptr);

  // x_new = x; x_new[mask] += attn_out  (atomic add: duplicate-index safe)
  k_copy_f32<<<2048, 256, 0, stream>>>(x, xnew, (long)SEQ * NE);
  k_scatter_add<<<(SEQ * NE + 255) / 256, 256, 0, stream>>>(attnf, mask, xnew, SEQ, NE);
}
